// RNNModel_27195732918387
// MI455X (gfx1250) — matrix operand order repaired and checked by reading, not compiled, not run on hardware
//
#include <hip/hip_runtime.h>
#include <hip/hip_bf16.h>

// ---------------------------------------------------------------------------
// Fused 2-layer GRU (B=1024,T=512,F=64,H=128) + gather-last + MLP head.
// Strategy: one workgroup (8 wave32) per 16 batch rows; weights pre-converted
// to f16 in WMMA B-fragment layout; Wg1/Wc1/Wg2 resident in LDS (272KB),
// Wc2 streamed from L2. All matmuls via v_wmma_f32_16x16x32_f16.
// ---------------------------------------------------------------------------

typedef __attribute__((ext_vector_type(16))) _Float16 v16h;
typedef __attribute__((ext_vector_type(8)))  _Float16 v8h;
typedef __attribute__((ext_vector_type(8)))  float    v8f;

#define B_   1024
#define T_   512
#define F_   64
#define H_   128

// ws (f16 halves) layout: fragment-major weights
#define OFF_WG1 0        // 192x256 -> 49152
#define OFF_WC1 49152    // 192x128 -> 24576
#define OFF_WG2 73728    // 256x256 -> 65536
#define OFF_WC2 139264   // 256x128 -> 32768
#define W_TOTAL 172032

// LDS layout (halves): Wg1|Wc1|Wg2 | A1 | A2 | H1 | H2 | U | (f32) Last | FC | seq
#define LDSW_HALVES 139264                 // Wg1+Wc1+Wg2
#define SMEM_BYTES  313920

union V16U { v16h v; v8h h[2]; };

// A fragment (16x32 f16, row-major source with row stride ldk halves).
// lane L: row = L&15; group = L>>4. halves[0..7] = K k0+grp*8.. ; [8..15] = K k0+16+grp*8..
__device__ __forceinline__ v16h load_a_frag(const _Float16* base, int ldk, int lane, int k0) {
  int row = lane & 15;
  int grp = lane >> 4;
  const _Float16* p = base + row * ldk + k0 + grp * 8;
  V16U u;
  u.h[0] = *(const v8h*)(p);
  u.h[1] = *(const v8h*)(p + 16);
  return u.v;
}

// B fragment from pre-swizzled fragment-major storage:
// tile (ktile,ntile) = 512 contiguous halves; lane owns 16 contiguous halves.
__device__ __forceinline__ v16h load_b_frag(const _Float16* base, int Nt, int ktile, int ntile, int lane) {
  const _Float16* p = base + (((ktile * Nt + ntile) << 9) + (lane << 4));
  V16U u;
  u.h[0] = *(const v8h*)(p);
  u.h[1] = *(const v8h*)(p + 8);
  return u.v;
}

__device__ __forceinline__ v8f wmma16(v16h a, v16h b, v8f c) {
  return __builtin_amdgcn_wmma_f32_16x16x32_f16(false, a, false, b, (short)0, c, false, false);
}

__device__ __forceinline__ v8f splat8(float b) {
  v8f r;
#pragma unroll
  for (int i = 0; i < 8; ++i) r[i] = b;
  return r;
}

__device__ __forceinline__ float sigm_(float x) { return 1.f / (1.f + __expf(-x)); }
__device__ __forceinline__ float tanh_(float x) {
  float e = __expf(-2.f * fabsf(x));
  float t = (1.f - e) / (1.f + e);
  return x < 0.f ? -t : t;
}

// ---------------------------------------------------------------------------
// Pre-pass: f32 row-major [K,N] weight -> f16 WMMA B-fragment layout.
// out index i: s=i&15 (K within 16), lane=(i>>4)&31, tile=i>>9 (ktile*Nt+ntile)
// ---------------------------------------------------------------------------
__global__ void gru_convert_frag_kernel(const float* __restrict__ src, _Float16* __restrict__ dst,
                                        int Nt, int N, int total) {
  int i = blockIdx.x * 256 + threadIdx.x;
  if (i >= total) return;
  int s    = i & 15;
  int lane = (i >> 4) & 31;
  int tile = i >> 9;
  int ntile = tile % Nt;
  int ktile = tile / Nt;
  int k = ktile * 32 + (lane >> 4) * 8 + (s & 7) + (s >> 3) * 16;
  int n = ntile * 16 + (lane & 15);
  dst[i] = (_Float16)src[k * N + n];
}

// ---------------------------------------------------------------------------
// Main fused kernel. grid = 64 workgroups, block = 256 (8 wave32).
// ---------------------------------------------------------------------------
__global__ __launch_bounds__(256, 1)
void gru_fused_kernel(const float* __restrict__ x, const int* __restrict__ seq,
                      const _Float16* __restrict__ wF,
                      const float* __restrict__ bg1, const float* __restrict__ bc1,
                      const float* __restrict__ bg2, const float* __restrict__ bc2,
                      const float* __restrict__ Wf1, const float* __restrict__ bf1,
                      const float* __restrict__ Wf2, const float* __restrict__ bf2,
                      float* __restrict__ out) {
  extern __shared__ char smem[];
  _Float16* sW    = (_Float16*)smem;
  _Float16* sWg1  = sW;                    // frag, Nt=16, Kt=6
  _Float16* sWc1  = sW + OFF_WC1;          // frag, Nt=8,  Kt=6
  _Float16* sWg2  = sW + OFF_WG2;          // frag, Nt=16, Kt=8
  _Float16* sA1   = sW + LDSW_HALVES;      // [16][192]
  _Float16* sA2   = sA1 + 16 * 192;        // [16][256]
  _Float16* sH1   = sA2 + 16 * 256;        // [16][128]
  _Float16* sH2   = sH1 + 16 * 128;        // [16][128]
  _Float16* sU    = sH2 + 16 * 128;        // [16][128]
  float*    sLast = (float*)(sU + 16 * 128);  // [16][128]
  float*    sFC   = sLast + 16 * 128;         // [16][8]
  int*      sSeq  = (int*)(sFC + 16 * 8);     // [16]

  const _Float16* gWc2 = wF + OFF_WC2;     // streamed from global/L2, Nt=8, Kt=8

  const int tid      = threadIdx.x;
  const int lane     = tid & 31;
  const int wave     = tid >> 5;
  const int row_base = blockIdx.x * 16;
  const int rbase    = (lane >> 4) * 8;    // D-layout row base for this half-wave

  // Stage resident weights (272KB) once: 17408 x 16B
  {
    const uint4* src = (const uint4*)wF;
    uint4*       dst = (uint4*)smem;
    for (int i = tid; i < (LDSW_HALVES * 2) / 16; i += 256) dst[i] = src[i];
  }
  if (tid < 16) sSeq[tid] = seq[row_base + tid];
  for (int i = tid; i < 16 * H_; i += 256) { sH1[i] = (_Float16)0.f; sH2[i] = (_Float16)0.f; }
  __syncthreads();

  // Per-wave column assignments + bias preload (all 8 D elems share one column)
  const int cg0 = wave * 32 + (lane & 15);       // gate tile 0 column
  const int cg1 = cg0 + 16;                      // gate tile 1 column
  const int nc  = wave * 16 + (lane & 15);       // candidate column
  const float bG1a = bg1[cg0], bG1b = bg1[cg1];
  const float bG2a = bg2[cg0], bG2b = bg2[cg1];
  const float bC1  = bc1[nc],  bC2  = bc2[nc];

  for (int t = 0; t < T_; ++t) {
    // ---- stage x_t (f32->f16) and h-state copies into A buffers ----
    for (int i = tid; i < 16 * F_; i += 256) {
      int r = i >> 6, f = i & 63;
      sA1[r * 192 + f] =
          (_Float16)x[(size_t)(row_base + r) * (T_ * F_) + (size_t)t * F_ + f];
    }
    for (int i = tid; i < 16 * H_; i += 256) {
      int r = i >> 7, c = i & 127;
      sA1[r * 192 + 64 + c]  = sH1[i];   // A1 = [x_t | h1_old]
      sA2[r * 256 + 128 + c] = sH2[i];   // A2 hi = h2_old (lo filled after combine1)
    }
    if (t + 1 < T_ && tid < 16)          // pull next step's x into cache early
      __builtin_prefetch(&x[(size_t)(row_base + tid) * (T_ * F_) + (size_t)(t + 1) * F_], 0, 0);
    __syncthreads();

    // ---- layer 1 gates: [16x192] @ Wg1[192x256] ----
    v8f g0 = splat8(bG1a), g1 = splat8(bG1b);
#pragma unroll
    for (int kt = 0; kt < 6; ++kt) {
      v16h a  = load_a_frag(sA1, 192, lane, kt * 32);
      v16h b0 = load_b_frag(sWg1, 16, kt, wave * 2, lane);
      v16h b1 = load_b_frag(sWg1, 16, kt, wave * 2 + 1, lane);
      g0 = wmma16(a, b0, g0);
      g1 = wmma16(a, b1, g1);
    }
    __syncthreads();  // everyone done reading A1 before r*h overwrite
#pragma unroll
    for (int j = 0; j < 8; ++j) {
      int r = rbase + j;
      float s0 = sigm_(g0[j]), s1 = sigm_(g1[j]);
      if (wave < 4) {  // r-gates (cols 0..127): A1 <- [x | r*h1_old]
        float h0 = (float)sH1[r * 128 + cg0];
        float h1v = (float)sH1[r * 128 + cg1];
        sA1[r * 192 + 64 + cg0] = (_Float16)(s0 * h0);
        sA1[r * 192 + 64 + cg1] = (_Float16)(s1 * h1v);
      } else {         // u-gates (cols 128..255)
        sU[r * 128 + (cg0 - 128)] = (_Float16)s0;
        sU[r * 128 + (cg1 - 128)] = (_Float16)s1;
      }
    }
    __syncthreads();

    // ---- layer 1 candidate + combine ----
    v8f cacc = splat8(bC1);
#pragma unroll
    for (int kt = 0; kt < 6; ++kt) {
      v16h a = load_a_frag(sA1, 192, lane, kt * 32);
      v16h b = load_b_frag(sWc1, 8, kt, wave, lane);
      cacc = wmma16(a, b, cacc);
    }
#pragma unroll
    for (int j = 0; j < 8; ++j) {   // each (row,col) owned by exactly one lane
      int r = rbase + j;
      float cv = tanh_(cacc[j]);
      float uv = (float)sU[r * 128 + nc];
      float ho = (float)sH1[r * 128 + nc];
      float hn = uv * ho + (1.f - uv) * cv;
      sH1[r * 128 + nc] = (_Float16)hn;
      sA2[r * 256 + nc] = (_Float16)hn;   // layer-2 input lo half
    }
    __syncthreads();

    // ---- layer 2 gates: [16x256] @ Wg2[256x256] ----
    g0 = splat8(bG2a); g1 = splat8(bG2b);
#pragma unroll
    for (int kt = 0; kt < 8; ++kt) {
      v16h a  = load_a_frag(sA2, 256, lane, kt * 32);
      v16h b0 = load_b_frag(sWg2, 16, kt, wave * 2, lane);
      v16h b1 = load_b_frag(sWg2, 16, kt, wave * 2 + 1, lane);
      g0 = wmma16(a, b0, g0);
      g1 = wmma16(a, b1, g1);
    }
    __syncthreads();
#pragma unroll
    for (int j = 0; j < 8; ++j) {
      int r = rbase + j;
      float s0 = sigm_(g0[j]), s1 = sigm_(g1[j]);
      if (wave < 4) {
        float h0 = (float)sH2[r * 128 + cg0];
        float h1v = (float)sH2[r * 128 + cg1];
        sA2[r * 256 + 128 + cg0] = (_Float16)(s0 * h0);
        sA2[r * 256 + 128 + cg1] = (_Float16)(s1 * h1v);
      } else {
        sU[r * 128 + (cg0 - 128)] = (_Float16)s0;
        sU[r * 128 + (cg1 - 128)] = (_Float16)s1;
      }
    }
    __syncthreads();

    // ---- layer 2 candidate (Wc2 streamed from L2) + combine + last-step gather ----
    cacc = splat8(bC2);
#pragma unroll
    for (int kt = 0; kt < 8; ++kt) {
      v16h a = load_a_frag(sA2, 256, lane, kt * 32);
      v16h b = load_b_frag(gWc2, 8, kt, wave, lane);
      cacc = wmma16(a, b, cacc);
    }
#pragma unroll
    for (int j = 0; j < 8; ++j) {
      int r = rbase + j;
      float cv = tanh_(cacc[j]);
      float uv = (float)sU[r * 128 + nc];
      float ho = (float)sH2[r * 128 + nc];
      float hn = uv * ho + (1.f - uv) * cv;
      sH2[r * 128 + nc] = (_Float16)hn;
      if (sSeq[r] == t + 1) sLast[r * 128 + nc] = hn;  // h2 at last valid step
    }
    __syncthreads();
  }

  // ---- head: relu(last) @ Wf1 + bf1 -> relu -> @ Wf2 + bf2 ----
  if (tid < 128) {
    int r = tid >> 3, j = tid & 7;
    float s = bf1[j];
#pragma unroll 8
    for (int k = 0; k < 128; ++k) {
      float z = sLast[r * 128 + k];
      z = z > 0.f ? z : 0.f;
      s += z * Wf1[k * 8 + j];
    }
    sFC[r * 8 + j] = s > 0.f ? s : 0.f;
  }
  __syncthreads();
  if (tid < 64) {
    int r = tid >> 2, c = tid & 3;
    float s = bf2[c];
#pragma unroll
    for (int k = 0; k < 8; ++k) s += sFC[r * 8 + k] * Wf2[k * 4 + c];
    out[(row_base + r) * 4 + c] = s;
  }
}

// ---------------------------------------------------------------------------
extern "C" void kernel_launch(void* const* d_in, const int* in_sizes, int n_in,
                              void* d_out, int out_size, void* d_ws, size_t ws_size,
                              hipStream_t stream) {
  const float* x    = (const float*)d_in[0];
  const int*   seq  = (const int*)d_in[1];
  const float* Wg1  = (const float*)d_in[2];
  const float* bg1  = (const float*)d_in[3];
  const float* Wc1  = (const float*)d_in[4];
  const float* bc1  = (const float*)d_in[5];
  const float* Wg2  = (const float*)d_in[6];
  const float* bg2  = (const float*)d_in[7];
  const float* Wc2  = (const float*)d_in[8];
  const float* bc2  = (const float*)d_in[9];
  const float* Wf1  = (const float*)d_in[10];
  const float* bf1  = (const float*)d_in[11];
  const float* Wf2  = (const float*)d_in[12];
  const float* bf2  = (const float*)d_in[13];
  float*       out  = (float*)d_out;
  _Float16*    wF   = (_Float16*)d_ws;

  (void)in_sizes; (void)n_in; (void)out_size; (void)ws_size;

  // Allow the large dynamic-LDS allocation (idempotent, capture-safe).
  (void)hipFuncSetAttribute((const void*)gru_fused_kernel,
                            hipFuncAttributeMaxDynamicSharedMemorySize, SMEM_BYTES);

  // f32 -> f16 fragment-layout weight conversion (tiny, once per call)
  gru_convert_frag_kernel<<<(49152 + 255) / 256, 256, 0, stream>>>(Wg1, wF + OFF_WG1, 16, 256, 49152);
  gru_convert_frag_kernel<<<(24576 + 255) / 256, 256, 0, stream>>>(Wc1, wF + OFF_WC1, 8, 128, 24576);
  gru_convert_frag_kernel<<<(65536 + 255) / 256, 256, 0, stream>>>(Wg2, wF + OFF_WG2, 16, 256, 65536);
  gru_convert_frag_kernel<<<(32768 + 255) / 256, 256, 0, stream>>>(Wc2, wF + OFF_WC2, 8, 128, 32768);

  gru_fused_kernel<<<B_ / 16, 256, SMEM_BYTES, stream>>>(
      x, seq, wF, bg1, bc1, bg2, bc2, Wf1, bf1, Wf2, bf2, out);
}
